// MultiAttention_60722247631706
// MI455X (gfx1250) — matrix operand order repaired and checked by reading, not compiled, not run on hardware
//
#include <hip/hip_runtime.h>
#include <hip/hip_bf16.h>
#include <cstdint>

#define D_MODELc 1024
#define N_HEADSc 16
#define N_KVc    4
#define HDc      64
#define N_REPc   4
#define WINDOWc  128
#define SCALEc   0.125f
#define Bc       2
#define Tc       2048

typedef __attribute__((ext_vector_type(16))) __bf16 v16bf;
typedef __attribute__((ext_vector_type(8)))  float  v8f;

union Frag {
  v16bf v;
  unsigned short s[16];
  uint4 q[2];
};

__device__ __forceinline__ unsigned short f2bf(float f) {
  unsigned int u = __float_as_uint(f);
  u += 0x7FFFu + ((u >> 16) & 1u);          // round-to-nearest-even
  return (unsigned short)(u >> 16);
}
__device__ __forceinline__ float bf2f(unsigned short h) {
  return __uint_as_float(((unsigned int)h) << 16);
}

// ---------------------------------------------------------------- cast f32->bf16 (x4 vectorized)
__global__ void cast_f32_bf16_x4(const float* __restrict__ in,
                                 unsigned short* __restrict__ out, int n4) {
  for (int i = blockIdx.x * blockDim.x + threadIdx.x; i < n4;
       i += gridDim.x * blockDim.x) {
    const float4 f = *(const float4*)(const void*)(in + 4 * (size_t)i);
    ushort4 o;
    o.x = f2bf(f.x); o.y = f2bf(f.y); o.z = f2bf(f.z); o.w = f2bf(f.w);
    *(ushort4*)(void*)(out + 4 * (size_t)i) = o;
  }
}

// ---------------------------------------------------------------- GEMM: C = A(M,K) * W(N,K)^T
// Double-buffered LDS pipeline; MODE 0: f32 out, 1: bf16 out, 2: SiLU->bf16 out
template <int MODE>
__global__ __launch_bounds__(256) void gemm_bf16_wmma(
    const unsigned short* __restrict__ A,
    const unsigned short* __restrict__ W,
    void* __restrict__ Cout, int M, int N, int K) {
  __shared__ unsigned short As[2][128 * 32];   // 2 x 8 KB
  __shared__ unsigned short Bs[2][128 * 32];   // 2 x 8 KB
  const int tid  = threadIdx.x;
  const int lane = tid & 31, wave = tid >> 5;
  const int wm = wave & 3, wn = wave >> 2;             // 4x2 wave grid -> 128x128 block tile
  const int ln = lane & 15, half = lane >> 4;
  const int m0 = blockIdx.y * 128, n0 = blockIdx.x * 128;
  const int lrow = tid >> 1, lhalf = tid & 1;          // cooperative tile loader

  const unsigned short* ga = A + (size_t)(m0 + lrow) * K + lhalf * 16;
  const unsigned short* gw = W + (size_t)(n0 + lrow) * K + lhalf * 16;
  unsigned short* la = As[0] + lrow * 32 + lhalf * 16;
  unsigned short* lb = Bs[0] + lrow * 32 + lhalf * 16;
  const int lds_stride = 128 * 32;                     // buffer 1 offset (ushorts)

  // prologue: stage slab 0 into buffer 0
  *(uint4*)(void*)(la) = *(const uint4*)(const void*)(ga);
  *(uint4*)(void*)(lb) = *(const uint4*)(const void*)(gw);
  __syncthreads();

  const int nstages = K >> 5;
  v8f acc[2][4] = {};
  for (int s = 0; s < nstages; ++s) {
    const int cur = s & 1, nxt = cur ^ 1;

    // issue next slab's global loads up front (overlap with WMMA below)
    uint4 ra, rb;
    const bool have_next = (s + 1 < nstages);
    if (have_next) {
      ra = *(const uint4*)(const void*)(ga + (size_t)(s + 1) * 32);
      rb = *(const uint4*)(const void*)(gw + (size_t)(s + 1) * 32);
      if (s + 2 < nstages) {                           // warm L2/WGP$ two slabs ahead
        __builtin_prefetch(ga + (size_t)(s + 2) * 32, 0, 3);
        __builtin_prefetch(gw + (size_t)(s + 2) * 32, 0, 3);
      }
    }

    // compute on current buffer
    Frag af[2], bf[4];
    const unsigned short* abuf = As[0] + cur * lds_stride;
    const unsigned short* bbuf = Bs[0] + cur * lds_stride;
#pragma unroll
    for (int ms = 0; ms < 2; ++ms) {                   // A frag: row=ln, K split by lane-half
      const unsigned short* ap = abuf + (wm * 32 + ms * 16 + ln) * 32;
      af[ms].q[0] = *(const uint4*)(const void*)(ap + half * 16);
      af[ms].q[1] = *(const uint4*)(const void*)(ap + half * 16 + 8);
    }
#pragma unroll
    for (int ns = 0; ns < 4; ++ns) {                   // B frag: col=ln, contiguous K per half
      const unsigned short* bp = bbuf + (wn * 64 + ns * 16 + ln) * 32 + half * 16;
      bf[ns].q[0] = *(const uint4*)(const void*)(bp);
      bf[ns].q[1] = *(const uint4*)(const void*)(bp + 8);
    }
#pragma unroll
    for (int ms = 0; ms < 2; ++ms)
#pragma unroll
      for (int ns = 0; ns < 4; ++ns)
        acc[ms][ns] = __builtin_amdgcn_wmma_f32_16x16x32_bf16(
            false, af[ms].v, false, bf[ns].v, (short)0, acc[ms][ns], false, false);

    // stage next slab into the other buffer (safe: nxt was last read before prev barrier)
    if (have_next) {
      *(uint4*)(void*)(la + nxt * lds_stride) = ra;
      *(uint4*)(void*)(lb + nxt * lds_stride) = rb;
    }
    __syncthreads();
  }

  // epilogue: fully unrolled so every vector extract is constant-indexed (no scratch spill)
  float* Cf = (float*)Cout;
  unsigned short* Cb = (unsigned short*)Cout;
#pragma unroll
  for (int ms = 0; ms < 2; ++ms)
#pragma unroll
    for (int ns = 0; ns < 4; ++ns) {
      const int row0 = m0 + wm * 32 + ms * 16 + (half ? 8 : 0);
      const int col  = n0 + wn * 64 + ns * 16 + ln;
#pragma unroll
      for (int r = 0; r < 8; ++r) {
        float v = acc[ms][ns][r];
        if (MODE == 0) {
          Cf[(size_t)(row0 + r) * N + col] = v;
        } else {
          if (MODE == 2) v = v / (1.0f + __expf(-v));  // SiLU
          Cb[(size_t)(row0 + r) * N + col] = f2bf(v);
        }
      }
    }
}

// ---------------------------------------------------------------- RoPE + [B,T,H*64] -> [B,H,T,64]
__global__ void rope_arrange(const unsigned short* __restrict__ lin,
                             unsigned short* __restrict__ out, int heads) {
  const int total = Bc * Tc * heads * (HDc / 2);
  for (int idx = blockIdx.x * blockDim.x + threadIdx.x; idx < total;
       idx += gridDim.x * blockDim.x) {
    const int i = idx & 31;
    const int h = (idx >> 5) % heads;
    const int t = (idx / (32 * heads)) % Tc;
    const int b = idx / (32 * heads * Tc);
    const int cols = heads * HDc;
    const unsigned short* row = lin + (size_t)(b * Tc + t) * cols + h * HDc;
    const float x1 = bf2f(row[i]);
    const float x2 = bf2f(row[i + 32]);
    const float inv_freq = __powf(10000.0f, -(float)i * (1.0f / 32.0f));
    const float ang = (float)t * inv_freq;
    const float c = cosf(ang), s = sinf(ang);
    unsigned short* orow = out + ((size_t)(b * heads + h) * Tc + t) * HDc;
    orow[i]      = f2bf(x1 * c - x2 * s);
    orow[i + 32] = f2bf(x1 * s + x2 * c);
  }
}

// ---------------------------------------------------------------- V: [B,T,4*64] -> [B,4,64,T]
__global__ void v_transpose(const unsigned short* __restrict__ lin,
                            unsigned short* __restrict__ out) {
  const int total = Bc * Tc * N_KVc * HDc;
  for (int idx = blockIdx.x * blockDim.x + threadIdx.x; idx < total;
       idx += gridDim.x * blockDim.x) {
    const int d  = idx & (HDc - 1);
    const int hk = (idx >> 6) & (N_KVc - 1);
    const int t  = (idx / (HDc * N_KVc)) % Tc;
    const int b  = idx / (HDc * N_KVc * Tc);
    out[((size_t)(b * N_KVc + hk) * HDc + d) * Tc + t] =
        lin[(size_t)(b * Tc + t) * (N_KVc * HDc) + hk * HDc + d];
  }
}

// ---------------------------------------------------------------- sliding-window attention
// 1 wave per (b, h, 16-query tile); 9 key tiles of 16 cover [q0-128, q0+16)
// Branch-free matrix phase: WMMA always runs with full EXEC (ISA requirement);
// out-of-range tiles use clamped addresses and are killed by the score mask.
__global__ __launch_bounds__(128) void attn_swin(
    const unsigned short* __restrict__ q_r,   // [B,16,T,64] bf16, roped
    const unsigned short* __restrict__ k_r,   // [B,4,T,64]  bf16, roped
    const unsigned short* __restrict__ v_t,   // [B,4,64,T]  bf16
    unsigned short* __restrict__ attn_out) {  // [B,T,1024]  bf16
  __shared__ float sc[4][16][160];
  const int lane = threadIdx.x & 31, wave = threadIdx.x >> 5;
  const int ln = lane & 15, half = lane >> 4;
  const int w   = blockIdx.x * 4 + wave;
  const int nqt = Tc / 16;
  const int qt  = w % nqt;
  const int h   = (w / nqt) % N_HEADSc;
  const int b   = w / (nqt * N_HEADSc);
  const int q0  = qt * 16;
  const int hk  = h / N_REPc;
  const int kstart = q0 - WINDOWc;

  // Q fragments (A-matrix, 2 chunks of K-dim 32 covering HD=64)
  Frag aq[2];
  {
    const unsigned short* qp = q_r + ((size_t)(b * N_HEADSc + h) * Tc + q0 + ln) * HDc;
#pragma unroll
    for (int c = 0; c < 2; ++c) {
      aq[c].q[0] = *(const uint4*)(const void*)(qp + c * 32 + half * 16);
      aq[c].q[1] = *(const uint4*)(const void*)(qp + c * 32 + half * 16 + 8);
    }
  }

  // S = Q K^T, masked + scaled into LDS (no control flow around WMMA)
  const unsigned short* kh = k_r + (size_t)(b * N_KVc + hk) * Tc * HDc;
  for (int kt = 0; kt < 9; ++kt) {
    const int kbase = kstart + kt * 16;
    const int kaddr = kbase < 0 ? 0 : kbase;  // clamped, 16-aligned, always in range
    const unsigned short* kp = kh + (size_t)(kaddr + ln) * HDc;
    Frag bk0, bk1;                            // both K chunks in flight before WMMA
    bk0.q[0] = *(const uint4*)(const void*)(kp + half * 16);
    bk0.q[1] = *(const uint4*)(const void*)(kp + half * 16 + 8);
    bk1.q[0] = *(const uint4*)(const void*)(kp + 32 + half * 16);
    bk1.q[1] = *(const uint4*)(const void*)(kp + 32 + half * 16 + 8);
    v8f s = {};
    s = __builtin_amdgcn_wmma_f32_16x16x32_bf16(
        false, aq[0].v, false, bk0.v, (short)0, s, false, false);
    s = __builtin_amdgcn_wmma_f32_16x16x32_bf16(
        false, aq[1].v, false, bk1.v, (short)0, s, false, false);
    const int j = kbase + ln;
#pragma unroll
    for (int r = 0; r < 8; ++r) {
      const int m = half ? r + 8 : r;
      const int i = q0 + m;
      const bool ok = (kbase >= 0) && (j <= i) && (j >= i - (WINDOWc - 1));
      sc[wave][m][kt * 16 + ln] = ok ? s[r] * SCALEc : -1e30f;
    }
  }
  // pad columns 144..159 so softmax drives them to prob 0
#pragma unroll
  for (int r = 0; r < 8; ++r) {
    const int m = half ? r + 8 : r;
    sc[wave][m][144 + ln] = -1e30f;
  }
  __syncthreads();

  // row softmax: all 32 lanes; lane (ln,half) owns row ln, cols [half*80, half*80+80)
  {
    float* row = sc[wave][ln] + half * 80;
    float mx = -1e30f;
    for (int c = 0; c < 80; ++c) { const float v = row[c]; mx = v > mx ? v : mx; }
    const float mo = __shfl_xor(mx, 16);
    mx = mo > mx ? mo : mx;
    float sum = 0.0f;
    for (int c = 0; c < 80; ++c) { const float e = __expf(row[c] - mx); row[c] = e; sum += e; }
    sum += __shfl_xor(sum, 16);
    const float inv = 1.0f / sum;
    for (int c = 0; c < 80; ++c) row[c] *= inv;
  }
  __syncthreads();

  // O = P V  (5 key chunks of 32 x 4 d-tiles of 16)
  v8f o[4] = {{}, {}, {}, {}};
  const unsigned short* vh = v_t + (size_t)(b * N_KVc + hk) * HDc * Tc;
  for (int kc = 0; kc < 5; ++kc) {
    Frag pf;
    const float* prow = sc[wave][ln];
    const int cb = kc * 32 + half * 16;
#pragma unroll
    for (int e = 0; e < 8; ++e) pf.s[e]     = f2bf(prow[cb + e]);
#pragma unroll
    for (int e = 0; e < 8; ++e) pf.s[8 + e] = f2bf(prow[cb + 8 + e]);
    int bt = kstart + kc * 32 + half * 16;        // 16-aligned; clamp OOB (prob 0 there)
    bt = bt < 0 ? 0 : (bt > Tc - 16 ? Tc - 16 : bt);
#pragma unroll
    for (int dn = 0; dn < 4; ++dn) {
      const unsigned short* vp = vh + (size_t)(dn * 16 + ln) * Tc + bt;
      Frag vf;
      vf.q[0] = *(const uint4*)(const void*)(vp);
      vf.q[1] = *(const uint4*)(const void*)(vp + 8);
      o[dn] = __builtin_amdgcn_wmma_f32_16x16x32_bf16(
          false, pf.v, false, vf.v, (short)0, o[dn], false, false);
    }
  }
#pragma unroll
  for (int dn = 0; dn < 4; ++dn)
#pragma unroll
    for (int r = 0; r < 8; ++r) {
      const int m = half ? r + 8 : r;
      attn_out[((size_t)(b * Tc) + q0 + m) * D_MODELc + h * HDc + dn * 16 + ln] =
          f2bf(o[dn][r]);
    }
}

// ---------------------------------------------------------------- host
extern "C" void kernel_launch(void* const* d_in, const int* in_sizes, int n_in,
                              void* d_out, int out_size, void* d_ws, size_t ws_size,
                              hipStream_t stream) {
  (void)in_sizes; (void)n_in; (void)out_size; (void)ws_size;
  const float* query = (const float*)d_in[0];
  const float* Wq    = (const float*)d_in[1];
  const float* Wk    = (const float*)d_in[2];
  const float* Wv    = (const float*)d_in[3];
  const float* W1    = (const float*)d_in[4];
  const float* W2    = (const float*)d_in[5];

  char* ws = (char*)d_ws;
  size_t off = 0;
  auto alloc = [&](size_t elems) -> unsigned short* {
    unsigned short* p = (unsigned short*)(ws + off);
    off += (elems * 2 + 255) & ~(size_t)255;
    return p;
  };
  const int NQ = Bc * Tc;                                   // 4096 token rows
  unsigned short* qbf  = alloc((size_t)NQ * D_MODELc);
  unsigned short* wqb  = alloc((size_t)D_MODELc * D_MODELc);
  unsigned short* wkb  = alloc((size_t)N_KVc * HDc * D_MODELc);
  unsigned short* wvb  = alloc((size_t)N_KVc * HDc * D_MODELc);
  unsigned short* w1b  = alloc((size_t)4 * D_MODELc * D_MODELc);
  unsigned short* w2b  = alloc((size_t)D_MODELc * 4 * D_MODELc);
  unsigned short* qlin = alloc((size_t)NQ * D_MODELc);
  unsigned short* klin = alloc((size_t)NQ * N_KVc * HDc);
  unsigned short* vlin = alloc((size_t)NQ * N_KVc * HDc);
  unsigned short* q_r  = alloc((size_t)NQ * D_MODELc);
  unsigned short* k_r  = alloc((size_t)NQ * N_KVc * HDc);
  unsigned short* v_t  = alloc((size_t)NQ * N_KVc * HDc);
  unsigned short* attn = alloc((size_t)NQ * D_MODELc);
  unsigned short* hbuf = alloc((size_t)NQ * 4 * D_MODELc);

  auto cast = [&](const float* src, unsigned short* dst, size_t n) {
    const size_t n4 = n / 4;                                // all sizes are multiples of 4
    int blocks = (int)((n4 + 255) / 256);
    if (blocks > 4096) blocks = 4096;
    cast_f32_bf16_x4<<<blocks, 256, 0, stream>>>(src, dst, (int)n4);
  };
  cast(query, qbf, (size_t)NQ * D_MODELc);
  cast(Wq, wqb, (size_t)D_MODELc * D_MODELc);
  cast(Wk, wkb, (size_t)N_KVc * HDc * D_MODELc);
  cast(Wv, wvb, (size_t)N_KVc * HDc * D_MODELc);
  cast(W1, w1b, (size_t)4 * D_MODELc * D_MODELc);
  cast(W2, w2b, (size_t)D_MODELc * 4 * D_MODELc);

  const dim3 blk(256);
  gemm_bf16_wmma<1><<<dim3(D_MODELc / 128, NQ / 128), blk, 0, stream>>>(
      qbf, wqb, qlin, NQ, D_MODELc, D_MODELc);
  gemm_bf16_wmma<1><<<dim3((N_KVc * HDc) / 128, NQ / 128), blk, 0, stream>>>(
      qbf, wkb, klin, NQ, N_KVc * HDc, D_MODELc);
  gemm_bf16_wmma<1><<<dim3((N_KVc * HDc) / 128, NQ / 128), blk, 0, stream>>>(
      qbf, wvb, vlin, NQ, N_KVc * HDc, D_MODELc);

  rope_arrange<<<2048, 256, 0, stream>>>(qlin, q_r, N_HEADSc);
  rope_arrange<<<512, 256, 0, stream>>>(klin, k_r, N_KVc);
  v_transpose<<<1024, 256, 0, stream>>>(vlin, v_t);

  attn_swin<<<(Bc * N_HEADSc * (Tc / 16)) / 4, 128, 0, stream>>>(q_r, k_r, v_t, attn);

  gemm_bf16_wmma<2><<<dim3((4 * D_MODELc) / 128, NQ / 128), blk, 0, stream>>>(
      attn, w1b, hbuf, NQ, 4 * D_MODELc, D_MODELc);
  gemm_bf16_wmma<0><<<dim3(D_MODELc / 128, NQ / 128), blk, 0, stream>>>(
      hbuf, w2b, (float*)d_out, NQ, D_MODELc, 4 * D_MODELc);
}
